// SuperBNN_41377714930100
// MI455X (gfx1250) — 8-bit matrix operand order repaired and checked by reading, compiled, not run on hardware
//
#include <hip/hip_runtime.h>
#include <cstdint>
#include <cstddef>


typedef int v8i __attribute__((ext_vector_type(8)));

static __host__ __device__ inline int ceil_div(int a, int b) { return (a + b - 1) / b; }

__global__ void k_quant_stem(const float* __restrict__ w, float* __restrict__ wq, int n) {
  __shared__ float red[256];
  float mx = 0.f;
  for (int i = threadIdx.x; i < n; i += 256) mx = fmaxf(mx, fabsf(tanhf(w[i])));
  red[threadIdx.x] = mx;
  __syncthreads();
  for (int off = 128; off; off >>= 1) {
    if (threadIdx.x < off) red[threadIdx.x] = fmaxf(red[threadIdx.x], red[threadIdx.x + off]);
    __syncthreads();
  }
  float m = red[0];
  for (int i = threadIdx.x; i < n; i += 256) {
    float t = tanhf(w[i]) / m;
    float v = (t + 1.f) * 0.5f * 255.f;
    float q = rintf(v) * (1.f / 255.f);
    wq[i] = 2.f * q - 1.f;
  }
}

__global__ void k_alpha(const float* __restrict__ w, float* __restrict__ alpha, int Kw) {
  __shared__ float red[256];
  int co = blockIdx.x;
  float s = 0.f;
  for (int i = threadIdx.x; i < Kw; i += 256) s += fabsf(w[(size_t)co * Kw + i]);
  red[threadIdx.x] = s;
  __syncthreads();
  for (int off = 128; off; off >>= 1) {
    if (threadIdx.x < off) red[threadIdx.x] += red[threadIdx.x + off];
    __syncthreads();
  }
  if (threadIdx.x == 0) alpha[co] = red[0] / (float)Kw;
}

template <int KS>
__global__ void k_prep_binw(const float* __restrict__ w, uint32_t* __restrict__ wpack,
                            int cinShift, int cout, int Ktrue) {
  int t = threadIdx.x;
  int lane = t >> 3;
  int r = t & 7;
  int col = lane & 15, half = lane >> 4;
  int co = blockIdx.x * 16 + col;
  int kc = blockIdx.y;
  int cin = 1 << cinShift;
  int k0 = kc * 64 + (r >> 2) * 32 + half * 16 + (r & 3) * 4;
  uint32_t v = 0;
  for (int j = 0; j < 4; ++j) {
    int kk = k0 + j;
    int8_t s = 0;
    if (kk < Ktrue && co < cout) {
      int f = kk >> cinShift;
      int ci = kk & (cin - 1);
      int ky = f / KS, kx = f - ky * KS;
      float wv = w[(((size_t)co * cin + ci) * KS + ky) * KS + kx];
      s = (wv >= 0.f) ? (int8_t)1 : (int8_t)-1;
    }
    v |= ((uint32_t)(uint8_t)s) << (8 * j);
  }
  wpack[(((size_t)blockIdx.x * gridDim.y + kc) * 32 + lane) * 8 + r] = v;
}

__global__ void k_binact(const float* __restrict__ a, const float* __restrict__ m,
                         uint32_t* __restrict__ q, size_t total4, int cmask) {
  size_t i = (size_t)blockIdx.x * blockDim.x + threadIdx.x;
  if (i >= total4) return;
  float4 v = ((const float4*)a)[i];
  int c0 = (int)((i * 4) & (size_t)cmask);
  float vs[4] = {v.x, v.y, v.z, v.w};
  uint32_t r = 0;
#pragma unroll
  for (int j = 0; j < 4; ++j) {
    int8_t s = (vs[j] + m[c0 + j]) >= 0.f ? (int8_t)1 : (int8_t)-1;
    r |= ((uint32_t)(uint8_t)s) << (8 * j);
  }
  q[i] = r;
}

template <int KS>
__global__ __launch_bounds__(128) void k_bconv(
    const int8_t* __restrict__ qin, const uint32_t* __restrict__ wpack,
    const float* __restrict__ alpha, float* __restrict__ z,
    float* __restrict__ statS, float* __restrict__ statQ,
    int M, int cinShift, int cout, int H, int W, int Ho, int Wo,
    int stride, int nK, int Ktrue) {
  const int PAD = KS / 2;
  __shared__ int8_t Alds[2][64][68];
  __shared__ float redS[32], redQ[32];

  int t = threadIdx.x;
  int lane = t & 31;
  int wave = t >> 5;
  int mblk = blockIdx.x * 64;
  int nt0 = blockIdx.y * 2;
  if (t < 32) { redS[t] = 0.f; redQ[t] = 0.f; }

  int lrow = t >> 1;
  int m = mblk + lrow;
  bool rowOK = (m < M);
  int HoWo = Ho * Wo;
  int mm = rowOK ? m : 0;
  int n = mm / HoWo;
  int rem = mm - n * HoWo;
  int oy = rem / Wo;
  int ox = rem - oy * Wo;
  int cin = 1 << cinShift;

  const uint32_t* wb0 = wpack + (size_t)nt0 * nK * 256 + (size_t)lane * 8;
  const uint32_t* wb1 = wb0 + (size_t)nK * 256;
  int mloc = wave * 16 + (lane & 15);
  int half = lane >> 4;

  auto loadA = [&](int kc, int buf) {
    uint32_t* arow = (uint32_t*)&Alds[buf][lrow][0];
#pragma unroll
    for (int i = 0; i < 8; ++i) {
      int kb = ((t & 1) * 8 + i) * 4;
      int k0 = kc * 64 + kb;
      uint32_t val = 0u;
      if (rowOK && k0 < Ktrue) {
        int f = k0 >> cinShift;
        int ci = k0 & (cin - 1);
        int ky = f / KS, kx = f - ky * KS;
        int iy = oy * stride + ky - PAD;
        int ix = ox * stride + kx - PAD;
        if (iy >= 0 && iy < H && ix >= 0 && ix < W)
          val = *(const uint32_t*)(qin + (((size_t)n * H + iy) * W + ix) * (size_t)cin + ci);
      }
      arow[kb >> 2] = val;
    }
  };

  v8i acc0 = {0, 0, 0, 0, 0, 0, 0, 0};
  v8i acc1 = {0, 0, 0, 0, 0, 0, 0, 0};

  loadA(0, 0);
  __syncthreads();

  for (int kc = 0; kc < nK; ++kc) {
    int cur = kc & 1;
    if (kc + 1 < nK) loadA(kc + 1, cur ^ 1);

    const uint32_t* wk0 = wb0 + (size_t)kc * 256;
    const uint32_t* wk1 = wb1 + (size_t)kc * 256;
    uint4 p0 = *(const uint4*)(wk0);
    uint4 p1 = *(const uint4*)(wk0 + 4);
    uint4 p2 = *(const uint4*)(wk1);
    uint4 p3 = *(const uint4*)(wk1 + 4);
    if (kc + 1 < nK) {
      __builtin_prefetch(wk0 + 256, 0, 1);
      __builtin_prefetch(wk1 + 256, 0, 1);
    }
    v8i bf0, bf1;
    bf0[0] = (int)p0.x; bf0[1] = (int)p0.y; bf0[2] = (int)p0.z; bf0[3] = (int)p0.w;
    bf0[4] = (int)p1.x; bf0[5] = (int)p1.y; bf0[6] = (int)p1.z; bf0[7] = (int)p1.w;
    bf1[0] = (int)p2.x; bf1[1] = (int)p2.y; bf1[2] = (int)p2.z; bf1[3] = (int)p2.w;
    bf1[4] = (int)p3.x; bf1[5] = (int)p3.y; bf1[6] = (int)p3.z; bf1[7] = (int)p3.w;

    const uint32_t* ar = (const uint32_t*)&Alds[cur][mloc][0];
    v8i afrag;
#pragma unroll
    for (int r = 0; r < 8; ++r) {
      int kb = (r >> 2) * 32 + half * 16 + (r & 3) * 4;
      afrag[r] = (int)ar[kb >> 2];
    }

    acc0 = __builtin_amdgcn_wmma_i32_16x16x64_iu8(true, afrag, true, bf0, acc0,
                                                  false, false);
    acc1 = __builtin_amdgcn_wmma_i32_16x16x64_iu8(true, afrag, true, bf1, acc1,
                                                  false, false);
    __syncthreads();
  }

  int col = lane & 15;
#pragma unroll
  for (int p = 0; p < 2; ++p) {
    v8i a = p ? acc1 : acc0;
    int co = nt0 * 16 + p * 16 + col;
    float al = alpha[co];
#pragma unroll
    for (int r = 0; r < 8; ++r) {
      int mo = mblk + wave * 16 + half * 8 + r;
      if (mo < M) {
        float v = al * (float)a[r];
        z[(size_t)mo * cout + co] = v;
        atomicAdd(&redS[p * 16 + col], v);
        atomicAdd(&redQ[p * 16 + col], v * v);
      }
    }
  }
  __syncthreads();
  if (t < 32) {
    atomicAdd(&statS[nt0 * 16 + t], redS[t]);
    atomicAdd(&statQ[nt0 * 16 + t], redQ[t]);
  }
}

__global__ __launch_bounds__(128) void k_stem_conv(
    const float* __restrict__ x, const float* __restrict__ wq,
    float* __restrict__ z, float* __restrict__ statS, float* __restrict__ statQ,
    int N, int H, int W, int Ho, int Wo) {
  __shared__ float redS[32], redQ[32];
  if (threadIdx.x < 32) { redS[threadIdx.x] = 0.f; redQ[threadIdx.x] = 0.f; }
  __syncthreads();
  int pix = blockIdx.x * 128 + threadIdx.x;
  int P = N * Ho * Wo;
  float acc[32];
#pragma unroll
  for (int c = 0; c < 32; ++c) acc[c] = 0.f;
  if (pix < P) {
    int HoWo = Ho * Wo;
    int n = pix / HoWo;
    int rem = pix - n * HoWo;
    int oy = rem / Wo;
    int ox = rem - oy * Wo;
    for (int ky = 0; ky < 3; ++ky) {
      int iy = oy * 2 + ky - 1;
      if (iy < 0 || iy >= H) continue;
      for (int kx = 0; kx < 3; ++kx) {
        int ix = ox * 2 + kx - 1;
        if (ix < 0 || ix >= W) continue;
        float xv0 = x[(((size_t)n * 3 + 0) * H + iy) * W + ix];
        float xv1 = x[(((size_t)n * 3 + 1) * H + iy) * W + ix];
        float xv2 = x[(((size_t)n * 3 + 2) * H + iy) * W + ix];
#pragma unroll
        for (int co = 0; co < 32; ++co) {
          const float* wp = wq + co * 27 + ky * 3 + kx;
          float a = acc[co];
          a = fmaf(xv0, wp[0], a);
          a = fmaf(xv1, wp[9], a);
          a = fmaf(xv2, wp[18], a);
          acc[co] = a;
        }
      }
    }
#pragma unroll
    for (int co = 0; co < 32; ++co) {
      z[(size_t)pix * 32 + co] = acc[co];
      atomicAdd(&redS[co], acc[co]);
      atomicAdd(&redQ[co], acc[co] * acc[co]);
    }
  }
  __syncthreads();
  if (threadIdx.x < 32) {
    atomicAdd(&statS[threadIdx.x], redS[threadIdx.x]);
    atomicAdd(&statQ[threadIdx.x], redQ[threadIdx.x]);
  }
}

__global__ void k_bn_finalize(const float* __restrict__ S, const float* __restrict__ Q,
                              const float* __restrict__ g, const float* __restrict__ b,
                              float* __restrict__ scale, float* __restrict__ shift,
                              int C, float invCnt) {
  int c = blockIdx.x * blockDim.x + threadIdx.x;
  if (c >= C) return;
  float mean = S[c] * invCnt;
  float var = Q[c] * invCnt - mean * mean;
  float rs = rsqrtf(var + 1e-5f);
  float sc = g[c] * rs;
  scale[c] = sc;
  shift[c] = b[c] - mean * sc;
}

__global__ void k_stem_post(float* __restrict__ z, const float* __restrict__ scale,
                            const float* __restrict__ shift, const float* __restrict__ m1,
                            const float* __restrict__ pa, const float* __restrict__ m2,
                            size_t total) {
  size_t i = (size_t)blockIdx.x * blockDim.x + threadIdx.x;
  if (i >= total) return;
  int c = (int)(i & 31);
  float o = scale[c] * z[i] + shift[c] + m1[c];
  o = o >= 0.f ? o : pa[c] * o;
  z[i] = o + m2[c];
}

__global__ void k_post1(float* __restrict__ z, const float* __restrict__ aIn,
                        const float* __restrict__ scale, const float* __restrict__ shift,
                        const float* __restrict__ m12, const float* __restrict__ pa,
                        const float* __restrict__ m13, size_t total, int cShift,
                        int stride, int H, int W, int Ho, int Wo) {
  size_t i = (size_t)blockIdx.x * blockDim.x + threadIdx.x;
  if (i >= total) return;
  int C = 1 << cShift;
  int c = (int)(i & (size_t)(C - 1));
  int mpix = (int)(i >> cShift);
  float o = scale[c] * z[i] + shift[c];
  float res;
  if (stride == 2) {
    int HoWo = Ho * Wo;
    int n = mpix / HoWo;
    int rem = mpix - n * HoWo;
    int oy = rem / Wo;
    int ox = rem - oy * Wo;
    size_t base = ((((size_t)n * H + oy * 2) * W + ox * 2) << cShift) + c;
    size_t rowst = (size_t)W << cShift;
    res = 0.25f * (aIn[base] + aIn[base + C] + aIn[base + rowst] + aIn[base + rowst + C]);
  } else {
    res = aIn[i];
  }
  o += res + m12[c];
  o = o >= 0.f ? o : pa[c] * o;
  z[i] = o + m13[c];
}

__global__ void k_post2(float* __restrict__ z, const float* __restrict__ o1,
                        const float* __restrict__ scale, const float* __restrict__ shift,
                        const float* __restrict__ m22, const float* __restrict__ pa,
                        const float* __restrict__ m23, size_t total, int coutShift,
                        int cinShift) {
  size_t i = (size_t)blockIdx.x * blockDim.x + threadIdx.x;
  if (i >= total) return;
  int cout = 1 << coutShift;
  int c = (int)(i & (size_t)(cout - 1));
  size_t mpix = i >> coutShift;
  float o = scale[c] * z[i] + shift[c];
  o += o1[(mpix << cinShift) + (c & ((1 << cinShift) - 1))];
  o += m22[c];
  o = o >= 0.f ? o : pa[c] * o;
  z[i] = o + m23[c];
}

__global__ void k_nhwc2nchw(const float* __restrict__ src, float* __restrict__ dst,
                            int C, int HW, size_t total) {
  size_t i = (size_t)blockIdx.x * blockDim.x + threadIdx.x;
  if (i >= total) return;
  int p = (int)(i % (size_t)HW);
  size_t t2 = i / (size_t)HW;
  int c = (int)(t2 % (size_t)C);
  int n = (int)(t2 / (size_t)C);
  dst[i] = src[((size_t)n * HW + p) * C + c];
}

extern "C" void kernel_launch(void* const* d_in, const int* in_sizes, int n_in,
                              void* d_out, int out_size, void* d_ws, size_t ws_size,
                              hipStream_t stream) {
  (void)in_sizes; (void)n_in; (void)out_size; (void)ws_size;
  const int N = 32;
  const float* x = (const float*)d_in[0];
  auto F = [&](int i) { return (const float*)d_in[i]; };

  char* ws = (char*)d_ws;
  size_t off = 0;
  auto take = [&](size_t bytes) -> char* {
    char* p = ws + off;
    off = (off + bytes + 255) & ~(size_t)255;
    return p;
  };
  float* STATS = (float*)take(11 * 1024 * sizeof(float));
  float* ALPHA = (float*)take(256 * sizeof(float));
  float* WQ = (float*)take(864 * sizeof(float));
  uint32_t* WPACK = (uint32_t*)take(640 * 1024);
  int8_t* Q1 = (int8_t*)take((size_t)32 * 112 * 112 * 32);
  int8_t* Q2 = (int8_t*)take((size_t)25088 * 128);
  float* Z1 = (float*)take((size_t)3211264 * sizeof(float));
  float* SLOT0 = (float*)take((size_t)12845056 * sizeof(float));
  float* SLOT1 = (float*)take((size_t)6422528 * sizeof(float));

  hipMemsetAsync(STATS, 0, 11 * 1024 * sizeof(float), stream);

  auto statBlk = [&](int id) { return STATS + (size_t)id * 1024; };

  {
    const float *w = F(1), *g = F(2), *b = F(3), *m1 = F(4), *pa = F(5), *m2 = F(6);
    int H = 224, W = 224, Ho = 112, Wo = 112;
    int P0 = N * Ho * Wo;
    float* S = statBlk(0);
    k_quant_stem<<<1, 256, 0, stream>>>(w, WQ, 864);
    k_stem_conv<<<ceil_div(P0, 128), 128, 0, stream>>>(x, WQ, SLOT0, S, S + 256, N, H, W, Ho, Wo);
    k_bn_finalize<<<1, 256, 0, stream>>>(S, S + 256, g, b, S + 512, S + 768, 32, 1.f / P0);
    size_t tot = (size_t)P0 * 32;
    k_stem_post<<<(int)((tot + 255) / 256), 256, 0, stream>>>(SLOT0, S + 512, S + 768, m1, pa, m2, tot);
  }

  const int cins[6] = {0, 32, 64, 128, 128, 256};
  const int couts[6] = {0, 64, 128, 128, 256, 256};
  const int Hins[6] = {0, 112, 56, 28, 28, 14};
  const int strides[6] = {0, 2, 2, 1, 2, 1};

  for (int s = 1; s <= 5; ++s) {
    int base = 7 + 14 * (s - 1);
    const float *m11 = F(base + 0), *w1 = F(base + 1), *g1 = F(base + 2), *b1 = F(base + 3);
    const float *m12 = F(base + 4), *pa1 = F(base + 5), *m13 = F(base + 6);
    const float *m21 = F(base + 7), *w2 = F(base + 8), *g2 = F(base + 9), *b2 = F(base + 10);
    const float *m22 = F(base + 11), *pa2 = F(base + 12), *m23 = F(base + 13);

    int cin = cins[s], cout = couts[s];
    int H = Hins[s], W = Hins[s], stride = strides[s];
    int Ho = (stride == 2) ? H / 2 : H, Wo = Ho;
    int Pin = N * H * W, Pout = N * Ho * Wo;
    int cinSh = __builtin_ctz(cin), coutSh = __builtin_ctz(cout);
    float* aIn = (s % 2 == 1) ? SLOT0 : SLOT1;
    float* aOut = (s % 2 == 1) ? SLOT1 : SLOT0;

    {
      int K = cin * 9, nK = ceil_div(K, 64);
      float* S = statBlk(2 * s - 1);
      k_alpha<<<cin, 256, 0, stream>>>(w1, ALPHA, K);
      k_prep_binw<3><<<dim3(cin / 16, nK), 256, 0, stream>>>(w1, WPACK, cinSh, cin, K);
      size_t t4 = (size_t)Pin * cin / 4;
      k_binact<<<(int)((t4 + 255) / 256), 256, 0, stream>>>(aIn, m11, (uint32_t*)Q1, t4, cin - 1);
      k_bconv<3><<<dim3(ceil_div(Pout, 64), cin / 32), 128, 0, stream>>>(
          Q1, WPACK, ALPHA, Z1, S, S + 256, Pout, cinSh, cin, H, W, Ho, Wo, stride, nK, K);
      k_bn_finalize<<<1, 256, 0, stream>>>(S, S + 256, g1, b1, S + 512, S + 768, cin, 1.f / Pout);
      size_t tot = (size_t)Pout * cin;
      k_post1<<<(int)((tot + 255) / 256), 256, 0, stream>>>(
          Z1, aIn, S + 512, S + 768, m12, pa1, m13, tot, cinSh, stride, H, W, Ho, Wo);
    }

    {
      int K = cin, nK = ceil_div(K, 64);
      float* S = statBlk(2 * s);
      k_alpha<<<cout, 256, 0, stream>>>(w2, ALPHA, K);
      k_prep_binw<1><<<dim3(cout / 16, nK), 256, 0, stream>>>(w2, WPACK, cinSh, cout, K);
      size_t t4 = (size_t)Pout * cin / 4;
      k_binact<<<(int)((t4 + 255) / 256), 256, 0, stream>>>(Z1, m21, (uint32_t*)Q2, t4, cin - 1);
      k_bconv<1><<<dim3(ceil_div(Pout, 64), cout / 32), 128, 0, stream>>>(
          Q2, WPACK, ALPHA, aOut, S, S + 256, Pout, cinSh, cout, Ho, Wo, Ho, Wo, 1, nK, K);
      k_bn_finalize<<<1, 256, 0, stream>>>(S, S + 256, g2, b2, S + 512, S + 768, cout, 1.f / Pout);
      size_t tot = (size_t)Pout * cout;
      k_post2<<<(int)((tot + 255) / 256), 256, 0, stream>>>(
          aOut, Z1, S + 512, S + 768, m22, pa2, m23, tot, coutSh, cinSh);
    }

    if (s == 1) {
      size_t tot = (size_t)N * 64 * 56 * 56;
      k_nhwc2nchw<<<(int)((tot + 255) / 256), 256, 0, stream>>>(aOut, (float*)d_out, 64, 56 * 56, tot);
    }
    if (s == 5) {
      size_t tot = (size_t)N * 256 * 14 * 14;
      k_nhwc2nchw<<<(int)((tot + 255) / 256), 256, 0, stream>>>(
          aOut, (float*)d_out + 6422528, 256, 14 * 14, tot);
    }
  }
}
